// NTXentLoss_86492051407169
// MI455X (gfx1250) — compile-verified, matrix operand order checked by reading, not run on hardware
//
#include <hip/hip_runtime.h>
#include <hip/hip_bf16.h>

typedef __attribute__((ext_vector_type(16))) __bf16 v16bf;
typedef __attribute__((ext_vector_type(8)))  __bf16 v8bf;
typedef __attribute__((ext_vector_type(8)))  float  v8f;
typedef __attribute__((ext_vector_type(4)))  float  v4f;

#define BATCH 4096
#define NROWS 8192
#define DIM   256
#define NTILE 512          // NROWS / 16
#define ROWP  (DIM + 8)    // padded LDS row: 528 B -> ds_load banks 4*lane mod 64
#define INV_T 2.0f         // 1 / TEMPERATURE(0.5)

// ---------------------------------------------------------------------------
// CDNA5 async global->LDS copy (ASYNCcnt path). The clang builtin exists but
// uses LangAS-qualified pointer types we cannot spell in HIP source, so issue
// the instruction with inline asm (VGLOBAL: VDST = LDS byte address VGPR,
// VADDR = 64-bit global address, no SADDR).
// ---------------------------------------------------------------------------
__device__ __forceinline__ void async_copy_b128(const __bf16* g, __bf16* l) {
#if defined(__AMDGCN__)
    const unsigned lds_off = (unsigned)(unsigned long long)
        (__attribute__((address_space(3))) const __bf16*)l;
    asm volatile("global_load_async_to_lds_b128 %0, %1, off"
                 :
                 : "v"(lds_off), "v"((unsigned long long)g)
                 : "memory");
#else
    *(v8bf*)l = *(const v8bf*)g;   // host-pass placeholder
#endif
}

__device__ __forceinline__ void wait_async_le2() {
#if defined(__AMDGCN__) && __has_builtin(__builtin_amdgcn_s_wait_asynccnt)
    __builtin_amdgcn_s_wait_asynccnt(2);
#elif defined(__AMDGCN__)
    asm volatile("s_wait_asynccnt 2" ::: "memory");
#endif
}

// ---------------------------------------------------------------------------
// Kernel 1: L2-normalize each of the 8192 rows (torch-style eps clamp) and
// store as bf16 row-major. One wave per row: 32 lanes x 8 floats = 256.
// ---------------------------------------------------------------------------
__global__ void __launch_bounds__(256)
ntxent_normalize(const float* __restrict__ zi, const float* __restrict__ zj,
                 __bf16* __restrict__ zn) {
    const int wave = (blockIdx.x * blockDim.x + threadIdx.x) >> 5;
    const int lane = threadIdx.x & 31;
    if (wave >= NROWS) return;
    const float* src = (wave < BATCH) ? (zi + (size_t)wave * DIM)
                                      : (zj + (size_t)(wave - BATCH) * DIM);
    v4f a = *(const v4f*)(src + lane * 8);
    v4f b = *(const v4f*)(src + lane * 8 + 4);
    float ss = 0.f;
#pragma unroll
    for (int i = 0; i < 4; ++i) ss += a[i] * a[i] + b[i] * b[i];
#pragma unroll
    for (int m = 16; m >= 1; m >>= 1) ss += __shfl_xor(ss, m, 32);
    const float scale = 1.0f / fmaxf(sqrtf(ss), 1e-8f);
    v8bf o;
#pragma unroll
    for (int i = 0; i < 4; ++i) {
        o[i]     = (__bf16)(a[i] * scale);
        o[i + 4] = (__bf16)(b[i] * scale);
    }
    *(v8bf*)(zn + (size_t)wave * DIM + lane * 8) = o;
}

// ---------------------------------------------------------------------------
// Kernel 2: fused  zn @ zn^T / T -> sum(exp) -> logsumexp + positive capture.
// One wave per 16-row M-tile; the 8 waves of a block SHARE each 8 KB B tile,
// staged in LDS by the async global->LDS engine with double buffering so the
// DMA for tile nt+1 overlaps WMMA+exp work on tile nt.
// Logits are bounded in [-2, 2] so sum(exp) needs no running max.
// ---------------------------------------------------------------------------
__global__ void __launch_bounds__(256)
ntxent_logits(const __bf16* __restrict__ zn, float* __restrict__ rowterm) {
    __shared__ __bf16 btile[2][16 * ROWP];   // 2 x 8.25 KB double buffer

    const int tid   = threadIdx.x;
    const int lane  = tid & 31;
    const int mtile = blockIdx.x * 8 + (tid >> 5);   // 0..511
    const int m0    = mtile * 16;
    const int khalf = lane >> 4;      // which K-half this lane holds
    const int nlan  = lane & 15;      // M index for A loads / N index for B & C
    const int Mrow  = khalf * 8;      // C rows held by this lane-half

    // A fragments: rows m0..m0+15, register-resident for the whole sweep.
    // 16-bit A 16x32 layout: lanes 0-15 hold K = kc*32 + {0..7, 16..23},
    // lanes 16-31 hold K = kc*32 + {8..15, 24..31}.
    const __bf16* arow = zn + (size_t)(m0 + nlan) * DIM;
    v16bf afrag[8];
#pragma unroll
    for (int kc = 0; kc < 8; ++kc) {
        const int kb = kc * 32 + khalf * 8;
        v8bf lo = *(const v8bf*)(arow + kb);
        v8bf hi = *(const v8bf*)(arow + kb + 16);
        afrag[kc] = __builtin_shufflevector(lo, hi,
            0, 1, 2, 3, 4, 5, 6, 7, 8, 9, 10, 11, 12, 13, 14, 15);
    }

    float ssum[8], pos[8];
#pragma unroll
    for (int v = 0; v < 8; ++v) { ssum[v] = 0.f; pos[v] = 0.f; }

    const int ptile = (mtile + 256) & (NTILE - 1);  // tile holding positive cols

    // Cooperative tile copy mapping: thread copies 32 B (16 bf16) of the tile.
    const int crow = tid >> 4;          // 0..15 : row within tile
    const int ccol = (tid & 15) * 16;   // 0..240: column start
    const int loff = crow * ROWP + ccol;

    // Prologue: DMA tile 0 into buffer 0.
    {
        const __bf16* g = zn + (size_t)crow * DIM + ccol;   // tile 0
        async_copy_b128(g,     &btile[0][loff]);
        async_copy_b128(g + 8, &btile[0][loff + 8]);
    }

    for (int nt = 0; nt < NTILE; ++nt) {
        const int cur = nt & 1;
        // Issue DMA for tile nt+1 (wraps to a harmless dummy on the last iter
        // so the wait immediate stays constant).
        {
            const int next = (nt + 1) & (NTILE - 1);
            const __bf16* g = zn + ((size_t)next * 16 + crow) * DIM + ccol;
            async_copy_b128(g,     &btile[cur ^ 1][loff]);
            async_copy_b128(g + 8, &btile[cur ^ 1][loff + 8]);
        }
        wait_async_le2();    // own tile-nt loads landed (4 outstanding -> <=2)
        __syncthreads();     // everyone's tile-nt loads landed

        // B fragment: column n == row (nt*16 + n) of Zn, read from LDS.
        const __bf16* brow = &btile[cur][nlan * ROWP];
        v8f c = {};
#pragma unroll
        for (int kc = 0; kc < 8; ++kc) {
            const int kb = kc * 32 + khalf * 8;
            v8bf lo = *(const v8bf*)(brow + kb);
            v8bf hi = *(const v8bf*)(brow + kb + 16);
            v16bf bfrag = __builtin_shufflevector(lo, hi,
                0, 1, 2, 3, 4, 5, 6, 7, 8, 9, 10, 11, 12, 13, 14, 15);
            c = __builtin_amdgcn_wmma_f32_16x16x32_bf16(
                    false, afrag[kc], false, bfrag, (short)0, c, false, false);
        }

        const bool hitTile = (nt == ptile);
#pragma unroll
        for (int v = 0; v < 8; ++v) {
            const float x = c[v] * INV_T;            // logit, in [-2, 2]
            ssum[v] += __expf(x);
            // positive pair sits on the diagonal of the partner tile:
            // C element (M, N) with N == M == Mrow + v lives on this lane.
            if (hitTile && (nlan == Mrow + v)) pos[v] = x;
        }
        __syncthreads();     // tile-nt buffer may be overwritten next iter
    }

    // Reduce across the 16-lane half holding each C row (masks stay in-half).
#pragma unroll
    for (int v = 0; v < 8; ++v) {
        float s = ssum[v], p = pos[v];
#pragma unroll
        for (int m = 8; m >= 1; m >>= 1) {
            s += __shfl_xor(s, m, 32);
            p += __shfl_xor(p, m, 32);   // exactly one lane contributed pos
        }
        ssum[v] = s; pos[v] = p;
    }
    if (nlan == 0) {
#pragma unroll
        for (int v = 0; v < 8; ++v)
            rowterm[m0 + Mrow + v] = __logf(ssum[v]) - pos[v];
    }
}

// ---------------------------------------------------------------------------
// Kernel 3: deterministic single-block reduction of the 8192 per-row terms.
// loss = sum_i (logsumexp_i - pos_i) / N
// ---------------------------------------------------------------------------
__global__ void __launch_bounds__(256)
ntxent_reduce(const float* __restrict__ rowterm, float* __restrict__ out) {
    __shared__ float sm[256];
    float acc = 0.f;
    for (int i = threadIdx.x; i < NROWS; i += 256) acc += rowterm[i];
    sm[threadIdx.x] = acc;
    __syncthreads();
    for (int s = 128; s > 0; s >>= 1) {
        if ((int)threadIdx.x < s) sm[threadIdx.x] += sm[threadIdx.x + s];
        __syncthreads();
    }
    if (threadIdx.x == 0) out[0] = sm[0] / (float)NROWS;
}

extern "C" void kernel_launch(void* const* d_in, const int* in_sizes, int n_in,
                              void* d_out, int out_size, void* d_ws, size_t ws_size,
                              hipStream_t stream) {
    const float* zi = (const float*)d_in[0];
    const float* zj = (const float*)d_in[1];

    __bf16* zn      = (__bf16*)d_ws;                                   // 4 MB
    float*  rowterm = (float*)((char*)d_ws +
                               (size_t)NROWS * DIM * sizeof(__bf16));  // 32 KB

    ntxent_normalize<<<NROWS / 8, 256, 0, stream>>>(zi, zj, zn);
    ntxent_logits<<<NTILE / 8, 256, 0, stream>>>(zn, rowterm);
    ntxent_reduce<<<1, 256, 0, stream>>>(rowterm, (float*)d_out);
}
